// MATnet_34342558499145
// MI455X (gfx1250) — compile-verified, matrix operand order checked by reading, not run on hardware
//
#include <hip/hip_runtime.h>
#include <hip/hip_bf16.h>

// ---------------------------------------------------------------------------
// MATnet forward on MI455X (gfx1250, wave32, WMMA).
// Big GEMMs use v_wmma_f32_16x16x32_bf16; softmax/reductions stay fp32.
// Round 2: N-register-blocked kemb GEMM (10 accumulators/wave, pre-packed
// bf16 Wf) and LDS-staged attmap GEMM (k-tile shared across 8 b's per block).
// ---------------------------------------------------------------------------

#define B_SZ   128
#define NQ     32
#define NW     12
#define NK     64
#define EMB    300
#define FEAT   2048
#define EMB_P  320          // EMB padded to multiple of 32 for bf16 WMMA K-dim
#define NPAD   320          // Wf rows padded to 20 N-tiles
#define EPS_C  1e-5f

typedef __attribute__((ext_vector_type(16))) __bf16 v16bf;
typedef __attribute__((ext_vector_type(8)))  __bf16 v8bf;
typedef __attribute__((ext_vector_type(8)))  float  v8f;

// ---------------------------------------------------------------------------
// Pack Wf (f32 [300][2048]) -> bf16 [320][2048], rows >= 300 zeroed.
// ---------------------------------------------------------------------------
__global__ void pack_wf(const float* __restrict__ Wf, __bf16* __restrict__ dst) {
    const int i = blockIdx.x * blockDim.x + threadIdx.x;
    if (i >= NPAD * FEAT) return;
    const int r = i / FEAT;
    dst[i] = (r < EMB) ? (__bf16)Wf[i] : (__bf16)0.0f;
}

// ---------------------------------------------------------------------------
// Kernel 1: k_emb[m][n] = sum_k feature[m][k]*Wf[n][k] + wv[label[m]][n] + bf[n]
// M = 8192 (512 tiles), N = 320 padded (2 halves x 10 tiles), K = 2048.
// One wave per (M-tile, N-half): A loaded/converted once, 10 WMMA accs.
// Feature is read only 2x total; Wf is bf16-packed and L2-resident.
// ---------------------------------------------------------------------------
__global__ void kemb_gemm(const float*  __restrict__ feature,
                          const __bf16* __restrict__ wf_bf,
                          const float*  __restrict__ wv,
                          const int*    __restrict__ label,
                          const float*  __restrict__ bfv,
                          float* __restrict__ kemb) {
    const int waveG = blockIdx.x * (blockDim.x >> 5) + (threadIdx.x >> 5); // 0..1023
    const int lane  = threadIdx.x & 31;
    const int mt    = waveG >> 1;       // 0..511
    const int half  = waveG & 1;        // N-half: tiles [half*10, half*10+10)
    const int kc    = (lane >> 4) * 8;  // per-lane K sub-chunk offset

    const float*  Arow  = feature + (size_t)(mt * 16 + (lane & 15)) * FEAT;
    const __bf16* Bbase = wf_bf + (size_t)(half * 160 + (lane & 15)) * FEAT;

    v8f acc[10];
#pragma unroll
    for (int j = 0; j < 10; ++j) acc[j] = (v8f){};

    for (int k0 = 0; k0 < FEAT; k0 += 32) {
        union { v16bf v; __bf16 e[16]; } ua;
#pragma unroll
        for (int j = 0; j < 8; ++j) {
            ua.e[j]     = (__bf16)Arow[k0 + kc + j];
            ua.e[8 + j] = (__bf16)Arow[k0 + 16 + kc + j];
        }
#pragma unroll
        for (int j = 0; j < 10; ++j) {
            union { v16bf v; v8bf h[2]; } ub;
            const __bf16* pB = Bbase + (size_t)j * 16 * FEAT + k0 + kc;
            ub.h[0] = *(const v8bf*)pB;
            ub.h[1] = *(const v8bf*)(pB + 16);
            acc[j] = __builtin_amdgcn_wmma_f32_16x16x32_bf16(
                false, ua.v, false, ub.v, (short)0, acc[j], false, false);
        }
    }

    const int mb = mt * 16 + (lane >> 4) * 8;
#pragma unroll
    for (int j = 0; j < 10; ++j) {
        const int n = (half * 10 + j) * 16 + (lane & 15);
        if (n < EMB) {
            const float bias = bfv[n];
#pragma unroll
            for (int i = 0; i < 8; ++i) {
                const int m = mb + i;
                kemb[(size_t)m * EMB + n] =
                    acc[j][i] + wv[(size_t)label[m] * EMB + n] + bias;
            }
        }
    }
}

// ---------------------------------------------------------------------------
// Generic pack: f32 [rows][scols] -> zero-padded bf16 [rows][EMB_P]
// ---------------------------------------------------------------------------
__global__ void pack_pad_bf16(const float* __restrict__ src,
                              __bf16* __restrict__ dst,
                              int rows, int scols) {
    const int i = blockIdx.x * blockDim.x + threadIdx.x;
    if (i >= rows * EMB_P) return;
    const int r = i / EMB_P, c = i % EMB_P;
    dst[i] = (c < scols) ? (__bf16)src[(size_t)r * scols + c] : (__bf16)0.0f;
}

// W_eff = Wp + EPS*Wm, packed to bf16 [304][320] (zero pad rows/cols >= 300)
__global__ void pack_weff(const float* __restrict__ Wp,
                          const float* __restrict__ Wm,
                          __bf16* __restrict__ dst) {
    const int i = blockIdx.x * blockDim.x + threadIdx.x;
    if (i >= 304 * EMB_P) return;
    const int r = i / EMB_P, c = i % EMB_P;
    float v = 0.0f;
    if (r < EMB && c < EMB) v = Wp[(size_t)r * EMB + c] + EPS_C * Wm[(size_t)r * EMB + c];
    dst[i] = (__bf16)v;
}

// ---------------------------------------------------------------------------
// Kernel 2: per-(b,q) word attention (fp32, softmax-dominated).
//   att[w][k] = <wv[query[b,q,w]], k_emb[b,k]> / sqrt(EMB)
//   q_max_att[w] = max_k softmax_k(att[w,:]) = 1 / sum_k exp(att[w,k]-max)
//   alpha = softmax_w(q_max_att); p_emb[b,q] = sum_w alpha[w]*q_emb[w]
// ---------------------------------------------------------------------------
__global__ void qatt_kernel(const int*   __restrict__ query,
                            const float* __restrict__ wv,
                            const float* __restrict__ kemb,
                            float* __restrict__ p_emb) {
    const int q = blockIdx.x;   // 0..31
    const int b = blockIdx.y;   // 0..127
    __shared__ float q_lds[NW * EMB];
    __shared__ float att[NW * NK];
    __shared__ float alpha[NW];
    __shared__ int   qidx[NW];
    const int tid = threadIdx.x;

    if (tid < NW) qidx[tid] = query[((size_t)b * NQ + q) * NW + tid];
    __syncthreads();

    for (int i = tid; i < NW * EMB; i += 256) {
        const int w = i / EMB, e = i % EMB;
        q_lds[i] = wv[(size_t)qidx[w] * EMB + e];
    }
    __syncthreads();

    const float scale = 0.057735026918962584f;   // 1/sqrt(300)
    const float* kb = kemb + (size_t)b * NK * EMB;
    for (int p = tid; p < NW * NK; p += 256) {
        const int w = p >> 6, k = p & 63;
        const float* qp = q_lds + w * EMB;
        const float* kp = kb + (size_t)k * EMB;
        float s = 0.f;
        for (int e = 0; e < EMB; ++e) s += qp[e] * kp[e];
        att[p] = s * scale;
    }
    __syncthreads();

    if (tid < NW) {
        float m = -3.4e38f;
        for (int k = 0; k < NK; ++k) m = fmaxf(m, att[tid * NK + k]);
        float Z = 0.f;
        for (int k = 0; k < NK; ++k) Z += __expf(att[tid * NK + k] - m);
        alpha[tid] = 1.0f / Z;                    // q_max_att[w]
    }
    __syncthreads();

    if (tid == 0) {                               // softmax over w (12 vals)
        float m = -3.4e38f;
        for (int w = 0; w < NW; ++w) m = fmaxf(m, alpha[w]);
        float Z = 0.f, e[NW];
        for (int w = 0; w < NW; ++w) { e[w] = __expf(alpha[w] - m); Z += e[w]; }
        for (int w = 0; w < NW; ++w) alpha[w] = e[w] / Z;
    }
    __syncthreads();

    float* pout = p_emb + ((size_t)b * NQ + q) * EMB;
    for (int e = tid; e < EMB; e += 256) {
        float s = 0.f;
#pragma unroll
        for (int w = 0; w < NW; ++w) s += alpha[w] * q_lds[w * EMB + e];
        pout[e] = s;
    }
}

// ---------------------------------------------------------------------------
// Kernel 3: p_proj = p_emb @ W_eff.T + (bp + EPS*bm)
// M = 4096 (256 tiles), N = 300 (19 tiles), K = 320 (10 steps), bf16 inputs.
// ---------------------------------------------------------------------------
__global__ void proj_gemm(const __bf16* __restrict__ pin,
                          const __bf16* __restrict__ weff,
                          const float*  __restrict__ bp,
                          const float*  __restrict__ bm,
                          float* __restrict__ p_proj) {
    const int waveG = blockIdx.x * (blockDim.x >> 5) + (threadIdx.x >> 5);
    const int lane  = threadIdx.x & 31;
    const int mt = waveG / 19;          // 0..255
    const int nt = waveG % 19;          // 0..18
    const int kc = (lane >> 4) * 8;

    const __bf16* pA = pin  + (size_t)(mt * 16 + (lane & 15)) * EMB_P;
    const __bf16* pB = weff + (size_t)(nt * 16 + (lane & 15)) * EMB_P;

    v8f acc = {};
#pragma unroll
    for (int k0 = 0; k0 < EMB_P; k0 += 32) {
        union { v16bf v; v8bf h[2]; } ua, ub;
        ua.h[0] = *(const v8bf*)(pA + k0 + kc);
        ua.h[1] = *(const v8bf*)(pA + k0 + 16 + kc);
        ub.h[0] = *(const v8bf*)(pB + k0 + kc);
        ub.h[1] = *(const v8bf*)(pB + k0 + 16 + kc);
        acc = __builtin_amdgcn_wmma_f32_16x16x32_bf16(
            false, ua.v, false, ub.v, (short)0, acc, false, false);
    }

    const int n = nt * 16 + (lane & 15);
    if (n < EMB) {
        const float bias = bp[n] + EPS_C * bm[n];
        const int mb = mt * 16 + (lane >> 4) * 8;
#pragma unroll
        for (int i = 0; i < 8; ++i)
            p_proj[(size_t)(mb + i) * EMB + n] = acc[i] + bias;
    }
}

// ---------------------------------------------------------------------------
// Kernel 4: cross-batch attmap + fused epilogue.
//   attmap[b,a,q,v] = <p_proj[b,q], k_emb[a,v]>
//   att_obj_sum[b,a,v] = sum_q softmax_v(attmap[b,a,q,:])[v]
//   logits[b,a] = (sum_q max_v attmap) / num_query[b];  target = eye(B)
// One block per (a, b-group of 8): k tile (64x320 bf16 = 40KB) staged in LDS
// once and reused for 8 b's; 8 waves each own one 16x16 WMMA tile.
// ---------------------------------------------------------------------------
__global__ void attmap_kernel(const __bf16* __restrict__ p_bf,
                              const __bf16* __restrict__ k_bf,
                              const int*    __restrict__ num_query,
                              float* __restrict__ out) {
    const int a  = blockIdx.x;   // k_emb batch index
    const int bg = blockIdx.y;   // group of 8 p_emb batch indices
    __shared__ __bf16 k_lds[NK * EMB_P];       // 40 KB
    __shared__ float att[NQ][NK];              //  8 KB
    __shared__ float sm_m[NQ], sm_Z[NQ];
    const int tid  = threadIdx.x;
    const int wave = tid >> 5, lane = tid & 31;
    const int mt = wave >> 2;           // 0..1  (q tile)
    const int nt = wave & 3;            // 0..3  (v tile)
    const int kc = (lane >> 4) * 8;

    // cooperative stage of k_bf[a] into LDS (16B chunks)
    {
        const v8bf* src = (const v8bf*)(k_bf + (size_t)a * NK * EMB_P);
        v8bf* dst = (v8bf*)k_lds;
        for (int i = tid; i < NK * EMB_P / 8; i += 256) dst[i] = src[i];
    }
    __syncthreads();

    const __bf16* kRow = k_lds + (size_t)(nt * 16 + (lane & 15)) * EMB_P;

    for (int bi = 0; bi < 8; ++bi) {
        const int b = bg * 8 + bi;
        const __bf16* pA = p_bf + (size_t)(b * NQ + mt * 16 + (lane & 15)) * EMB_P;

        v8f acc = {};
#pragma unroll
        for (int k0 = 0; k0 < EMB_P; k0 += 32) {
            union { v16bf v; v8bf h[2]; } ua, ub;
            ua.h[0] = *(const v8bf*)(pA + k0 + kc);
            ua.h[1] = *(const v8bf*)(pA + k0 + 16 + kc);
            ub.h[0] = *(const v8bf*)(kRow + k0 + kc);
            ub.h[1] = *(const v8bf*)(kRow + k0 + 16 + kc);
            acc = __builtin_amdgcn_wmma_f32_16x16x32_bf16(
                false, ua.v, false, ub.v, (short)0, acc, false, false);
        }

        const int n  = nt * 16 + (lane & 15);
        const int mb = mt * 16 + (lane >> 4) * 8;
#pragma unroll
        for (int i = 0; i < 8; ++i) att[mb + i][n] = acc[i];
        __syncthreads();

        if (tid < NQ) {                 // per-q row stats over v
            float m = -3.4e38f;
            for (int v = 0; v < NK; ++v) m = fmaxf(m, att[tid][v]);
            float Z = 0.f;
            for (int v = 0; v < NK; ++v) Z += __expf(att[tid][v] - m);
            sm_m[tid] = m; sm_Z[tid] = Z;
        }
        __syncthreads();

        if (tid == 0) {
            float s = 0.f;
            for (int qq = 0; qq < NQ; ++qq) s += sm_m[qq];
            out[(size_t)b * B_SZ + a] = s / (float)num_query[b];            // logits
            out[B_SZ * B_SZ + (size_t)b * B_SZ + a] = (a == b) ? 1.f : 0.f; // target
        }
        if (tid < NK) {                 // column sums of row-softmaxed map
            float s = 0.f;
            for (int qq = 0; qq < NQ; ++qq)
                s += __expf(att[qq][tid] - sm_m[qq]) / sm_Z[qq];
            out[2 * B_SZ * B_SZ + ((size_t)b * B_SZ + a) * NK + tid] = s;   // att_obj_sum
        }
        __syncthreads();                // protect att/sm before next bi
    }
}

// ---------------------------------------------------------------------------
// Host-side launcher
// ---------------------------------------------------------------------------
extern "C" void kernel_launch(void* const* d_in, const int* in_sizes, int n_in,
                              void* d_out, int out_size, void* d_ws, size_t ws_size,
                              hipStream_t stream) {
    // inputs (setup_inputs dict order)
    const int*   query   = (const int*)  d_in[1];
    const int*   label   = (const int*)  d_in[2];
    const float* feature = (const float*)d_in[3];
    const int*   num_q   = (const int*)  d_in[7];
    const float* wv      = (const float*)d_in[8];
    const float* Wp      = (const float*)d_in[9];
    const float* bp      = (const float*)d_in[10];
    const float* Wm      = (const float*)d_in[11];
    const float* bm      = (const float*)d_in[12];
    const float* Wf      = (const float*)d_in[13];
    const float* bfv     = (const float*)d_in[14];
    float* out = (float*)d_out;

    // workspace layout (all offsets 16B aligned)
    char* ws = (char*)d_ws;
    float*  kemb   = (float*) (ws + 0);                         //  8192*300*4
    float*  p_emb  = (float*) (ws + 9830400);                   //  4096*300*4
    float*  p_proj = (float*) (ws + 14745600);                  //  4096*300*4
    __bf16* k_bf   = (__bf16*)(ws + 19660800);                  //  8192*320*2
    __bf16* pin_bf = (__bf16*)(ws + 24903680);                  //  4096*320*2
    __bf16* p_bf   = (__bf16*)(ws + 27525120);                  //  4096*320*2
    __bf16* weff   = (__bf16*)(ws + 30146560);                  //   304*320*2
    __bf16* wf_bf  = (__bf16*)(ws + 30341120);                  //   320*2048*2

    // 0) pack Wf -> bf16 [320][2048]
    pack_wf<<<(NPAD * FEAT + 255) / 256, 256, 0, stream>>>(Wf, wf_bf);

    // 1) k_emb = feature @ Wf.T + wv[label] + bf  (512 M-tiles x 2 N-halves)
    kemb_gemm<<<128, 256, 0, stream>>>(feature, wf_bf, wv, label, bfv, kemb);

    // 2) pack k_emb -> bf16 [8192][320]
    pack_pad_bf16<<<(8192 * EMB_P + 255) / 256, 256, 0, stream>>>(kemb, k_bf, 8192, EMB);

    // 3) word attention -> p_emb
    qatt_kernel<<<dim3(NQ, B_SZ), 256, 0, stream>>>(query, wv, kemb, p_emb);

    // 4) pack p_emb -> bf16, pack W_eff
    pack_pad_bf16<<<(4096 * EMB_P + 255) / 256, 256, 0, stream>>>(p_emb, pin_bf, 4096, EMB);
    pack_weff<<<(304 * EMB_P + 255) / 256, 256, 0, stream>>>(Wp, Wm, weff);

    // 5) projection GEMM (256*19 waves)
    proj_gemm<<<608, 256, 0, stream>>>(pin_bf, weff, bp, bm, p_proj);

    // 6) pack p_proj -> bf16
    pack_pad_bf16<<<(4096 * EMB_P + 255) / 256, 256, 0, stream>>>(p_proj, p_bf, 4096, EMB);

    // 7) cross-batch attmap + logits/target/att_obj_sum (128 a x 16 b-groups)
    attmap_kernel<<<dim3(B_SZ, B_SZ / 8), 256, 0, stream>>>(p_bf, k_bf, num_q, out);
}
